// Model_68410239091457
// MI455X (gfx1250) — matrix operand order repaired and checked by reading, not compiled, not run on hardware
//
#include <hip/hip_runtime.h>
#include <math.h>

// ============================================================================
// TimeXer-style forecaster forward for MI455X (gfx1250, wave32).
// - All dense matmuls: v_wmma_f32_16x16x32_f16 (f32 accumulate).
// - Tiles staged LDS via the Tensor Data Mover (tensor_load_to_lds, double
//   buffered, s_wait_tensorcnt), operands pre-packed f32->f16 with weights
//   transposed to [N][K] so every WMMA fragment is a contiguous LDS read.
// - DFT(len 307, prime) = 2 batched GEMMs vs cos/-sin matrices.
// - The fwm/fwp + fmm/fmp FFT filter collapses algebraically to an
//   elementwise scale by sigmoid(m)*cos(phi) (same constant on every bin).
// ============================================================================

typedef __attribute__((ext_vector_type(16))) _Float16 v16h;
typedef __attribute__((ext_vector_type(8)))  _Float16 v8h;
typedef __attribute__((ext_vector_type(8)))  float    v8f;
typedef __attribute__((ext_vector_type(4)))  unsigned int v4u;
typedef __attribute__((ext_vector_type(8)))  int      v8i;
typedef __attribute__((ext_vector_type(4)))  int      v4i;

#define BATCH  16
#define LSEQ   365
#define NCH    301
#define DM     512
#define DFFN   2048
#define NHEAD  8
#define LTOK   307
#define LENC   308
#define PLEN   7

__device__ __forceinline__ float sigm(float x) { return 1.f / (1.f + expf(-x)); }

// ---------------------------------------------------------------------------
// TDM 2-D tile load: global (f16, row-major, row stride `stride0` elements)
// -> LDS (rows packed back to back). D# per ISA ch.8:
//  group0: [1:0]=count=1, [63:32]=lds_addr(bytes), [120:64]=global_addr,
//          [127:126]=type=2
//  group1: [17:16]=data_size(1 => 2B), [79:48]=tensor_dim0(remaining),
//          [111:80]=tensor_dim1(remaining), [127:112]=tile_dim0,
//          [143:128]=tile_dim1, [207:160]=tensor_dim0_stride
// Tensor dims are the remaining extents measured from the tile start, so the
// TDM zero-fills K/M tails for us. This toolchain's builtin is the 6-arg
// (clang-23 / therock) form: (v4u, v8i, v4i, v4i, v8i, i32 cpol).
// ---------------------------------------------------------------------------
__device__ __forceinline__ void tdm_load_2d(unsigned lds_off, const _Float16* gptr,
                                            unsigned rem0, unsigned rem1,
                                            unsigned tile0, unsigned tile1,
                                            unsigned stride0)
{
    unsigned long long ga = (unsigned long long)(uintptr_t)gptr;
    v4u g0;
    g0.x = 1u;                                   // count=1, user mode
    g0.y = lds_off;                              // LDS byte address
    g0.z = (unsigned)(ga & 0xFFFFFFFFu);
    g0.w = (unsigned)((ga >> 32) & 0x01FFFFFFu) | 0x80000000u; // type=2
    v8i g1;
    g1[0] = (int)(1u << 16);                     // data_size = 2 bytes
    g1[1] = (int)((rem0 & 0xFFFFu) << 16);
    g1[2] = (int)((rem0 >> 16) | ((rem1 & 0xFFFFu) << 16));
    g1[3] = (int)((rem1 >> 16) | (tile0 << 16));
    g1[4] = (int)(tile1 & 0xFFFFu);              // tile_dim1 (tile_dim2 = 0)
    g1[5] = (int)stride0;                        // stride low 32
    g1[6] = 0;                                   // stride hi / dim1_stride
    g1[7] = 0;
    v4i z4 = { 0, 0, 0, 0 };
    v8i z8 = { 0, 0, 0, 0, 0, 0, 0, 0 };
    __builtin_amdgcn_tensor_load_to_lds(g0, g1, z4, z4, z8, 0);
}

// ---------------------------------------------------------------------------
// WMMA GEMM:  C[m,n] = act( A[M,K] @ W[K,N] + bias[n] )
// A16: f16 row-major [M,K].  Wt16: f16 TRANSPOSED [N,K].
// Block tile 128x64, K-step 32, 8 waves; wave w owns rows w*16..w*16+15 and
// all 64 N (4 accumulators). TDM double-buffers the tiles; every fragment is
// a contiguous LDS read (A: 2 x b128, B: 1 x 32B).
// ---------------------------------------------------------------------------
__launch_bounds__(256)
__global__ void gemm_wmma(const _Float16* __restrict__ A16, long sA,
                          const _Float16* __restrict__ Wt16, long sW,
                          const float* __restrict__ bias,
                          float* __restrict__ C, long sC, int ldC,
                          int M, int N, int K, int act)
{
    __shared__ _Float16 As[2][128 * 32] __attribute__((aligned(128)));
    __shared__ _Float16 Bs[2][64 * 32]  __attribute__((aligned(128)));

    const _Float16* A = A16 + (long)blockIdx.z * sA;
    const _Float16* W = Wt16 + (long)blockIdx.z * sW;
    C += (long)blockIdx.z * sC;

    const int tid  = threadIdx.x;
    const int wv   = tid >> 5;
    const int lane = tid & 31;
    const int m0   = blockIdx.x * 128;
    const int n0   = blockIdx.y * 64;
    const int numK = (K + 31) >> 5;

    v8f acc[4];
    for (int nt = 0; nt < 4; ++nt) {
        float b = bias ? bias[n0 + nt * 16 + (lane & 15)] : 0.f;
        for (int r = 0; r < 8; ++r) acc[nt][r] = b;
    }

    const bool wave0 = (tid < 32);
    if (wave0) {
        tdm_load_2d((unsigned)(uintptr_t)&As[0][0], A + (long)m0 * K,
                    (unsigned)K, (unsigned)(M - m0), 32u, 128u, (unsigned)K);
        tdm_load_2d((unsigned)(uintptr_t)&Bs[0][0], W + (long)n0 * K,
                    (unsigned)K, (unsigned)(N - n0), 32u, 64u, (unsigned)K);
    }

    for (int kt = 0; kt < numK; ++kt) {
        const int cur = kt & 1;
        if (wave0) {
            if (kt + 1 < numK) {
                const int k1 = (kt + 1) << 5;
                tdm_load_2d((unsigned)(uintptr_t)&As[cur ^ 1][0],
                            A + (long)m0 * K + k1,
                            (unsigned)(K - k1), (unsigned)(M - m0),
                            32u, 128u, (unsigned)K);
                tdm_load_2d((unsigned)(uintptr_t)&Bs[cur ^ 1][0],
                            W + (long)n0 * K + k1,
                            (unsigned)(K - k1), (unsigned)(N - n0),
                            32u, 64u, (unsigned)K);
                __builtin_amdgcn_s_wait_tensorcnt(2); // tile kt's pair done
            } else {
                __builtin_amdgcn_s_wait_tensorcnt(0);
            }
        }
        __syncthreads();

        // A fragment (16x32 f16): lane m = lane&15, kh = lane>>4;
        // elems j -> K = j + 16*kh (matches B).
        const _Float16* ap = &As[cur][(wv * 16 + (lane & 15)) * 32 + ((lane >> 4) << 4)];
        union { v16h v; v8h h[2]; } a;
        a.h[0] = *(const v8h*)ap;
        a.h[1] = *(const v8h*)(ap + 8);

        for (int nt = 0; nt < 4; ++nt) {
            // B fragment (32x16): lane n = lane&15, kh = lane>>4;
            // elems j -> K = j + 16*kh  => contiguous 16 halfs in [N,K] tile.
            const _Float16* bp = &Bs[cur][(nt * 16 + (lane & 15)) * 32 + ((lane >> 4) << 4)];
            v16h b = *(const v16h*)bp;
            acc[nt] = __builtin_amdgcn_wmma_f32_16x16x32_f16(
                false, a.v, false, b, (short)0, acc[nt], false, false);
        }
        __syncthreads();
    }

    // Epilogue. C/D layout: elem r -> M = r + 8*(lane>>4), N = lane&15.
    for (int nt = 0; nt < 4; ++nt) {
        for (int r = 0; r < 8; ++r) {
            int gm = m0 + wv * 16 + r + ((lane >> 4) << 3);
            if (gm >= M) continue;
            int gn = n0 + nt * 16 + (lane & 15);
            float v = acc[nt][r];
            if (act == 1) v = 0.5f * v * (1.f + erff(v * 0.70710678118f)); // exact GELU
            C[(long)gm * ldC + gn] = v;
        }
    }
}

// f32 -> f16 row-major pack (activations).
__global__ void pack_rm(const float* __restrict__ in, _Float16* __restrict__ out, long total)
{
    long i = (long)blockIdx.x * blockDim.x + threadIdx.x;
    if (i < total) out[i] = (_Float16)in[i];
}

// f32 [K,N] -> f16 [N,K] transpose pack (weights / DFT activations), batched.
__global__ void pack_tr(const float* __restrict__ in, _Float16* __restrict__ out,
                        int K, int N, long sIn, long sOut, long total)
{
    long i = (long)blockIdx.x * blockDim.x + threadIdx.x;
    if (i >= total) return;
    long per = (long)K * N;
    long b = i / per, r = i % per;
    int n = (int)(r / K), k = (int)(r % K);
    out[b * sOut + r] = (_Float16)in[b * sIn + (long)k * N + n];
}

// ---------------------------------------------------------------------------
// Flow attention: one workgroup per (b,h). q,k,v tiles in dynamic LDS
// (~253KB for L=308; CDNA5 WGP has 320KB).
// ---------------------------------------------------------------------------
__launch_bounds__(256)
__global__ void flow_attn(const float* __restrict__ Q, const float* __restrict__ K,
                          const float* __restrict__ V, float* __restrict__ O, int L)
{
    extern __shared__ float sm[];
    const int b = blockIdx.x / NHEAD, h = blockIdx.x % NHEAD;
    float* q    = sm;
    float* k    = q + L * 64;
    float* v    = k + L * 64;
    float* kv   = v + L * 64;           // 64x64
    float* nr   = kv + 64 * 64;
    float* nc   = nr + L;
    float* cr   = nc + L;
    float* rr   = cr + L;
    float* ksum = rr + L;
    float* qsum = ksum + 64;
    float* kn   = qsum + 64;
    float* qn   = kn + 64;
    float* red  = qn + 64;

    const int tid = threadIdx.x;
    const float eps = 1e-6f;
    const long base = ((long)b * L) * DM + h * 64;

    for (int i = tid; i < L * 64; i += 256) {
        int l = i >> 6, e = i & 63;
        long gi = base + (long)l * DM + e;
        q[i] = sigm(Q[gi]);
        k[i] = sigm(K[gi]);
        v[i] = V[gi];
    }
    __syncthreads();
    if (tid < 64) {
        float s1 = 0.f, s2 = 0.f;
        for (int l = 0; l < L; ++l) { s1 += k[l * 64 + tid]; s2 += q[l * 64 + tid]; }
        ksum[tid] = s1; qsum[tid] = s2;
    }
    __syncthreads();
    for (int l = tid; l < L; l += 256) {
        float a = 0.f, c2 = 0.f;
        for (int d = 0; d < 64; ++d) {
            a  += (q[l * 64 + d] + eps) * (ksum[d] + eps);
            c2 += (k[l * 64 + d] + eps) * (qsum[d] + eps);
        }
        nr[l] = 1.f / a; nc[l] = 1.f / c2;
    }
    __syncthreads();
    if (tid < 64) {
        float a = 0.f, c2 = 0.f;
        for (int l = 0; l < L; ++l) { a += k[l * 64 + tid] * nc[l]; c2 += q[l * 64 + tid] * nr[l]; }
        kn[tid] = a; qn[tid] = c2;
    }
    __syncthreads();
    for (int l = tid; l < L; l += 256) {
        float a = 0.f, c2 = 0.f;
        for (int d = 0; d < 64; ++d) {
            a  += (q[l * 64 + d] + eps) * (kn[d] + eps);
            c2 += (k[l * 64 + d] + eps) * (qn[d] + eps);
        }
        rr[l] = sigm(a);
        cr[l] = c2;
    }
    __syncthreads();
    if (tid == 0) {
        float m = -1e30f;
        for (int l = 0; l < L; ++l) m = fmaxf(m, cr[l]);
        float s = 0.f;
        for (int l = 0; l < L; ++l) s += expf(cr[l] - m);
        red[0] = m; red[1] = s;
    }
    __syncthreads();
    float mm = red[0], ssum = red[1];
    for (int l = tid; l < L; l += 256) cr[l] = expf(cr[l] - mm) / ssum * (float)L;
    __syncthreads();
    for (int i = tid; i < 64 * 64; i += 256) {
        int d = i >> 6, e = i & 63;
        float a = 0.f;
        for (int l = 0; l < L; ++l) a += k[l * 64 + d] * v[l * 64 + e] * cr[l];
        kv[i] = a;
    }
    __syncthreads();
    for (int i = tid; i < L * 64; i += 256) {
        int l = i >> 6, e = i & 63;
        float a = 0.f;
        for (int d = 0; d < 64; ++d) a += q[l * 64 + d] * kv[d * 64 + e];
        O[base + (long)l * DM + e] = a * nr[l] * rr[l];
    }
}

// LayerNorm over D=512 with optional fused residual; in-place safe.
__launch_bounds__(256)
__global__ void ln_kernel(const float* __restrict__ x, const float* __restrict__ res,
                          const float* __restrict__ g, const float* __restrict__ bt,
                          float* __restrict__ out)
{
    __shared__ float rs[256], rs2[256];
    const long row = blockIdx.x;
    const float* xp = x + row * DM;
    const float* rp = res ? res + row * DM : nullptr;
    const int tid = threadIdx.x;
    float v0 = xp[tid]       + (rp ? rp[tid]       : 0.f);
    float v1 = xp[tid + 256] + (rp ? rp[tid + 256] : 0.f);
    rs[tid]  = v0 + v1;
    rs2[tid] = v0 * v0 + v1 * v1;
    __syncthreads();
    for (int s = 128; s > 0; s >>= 1) {
        if (tid < s) { rs[tid] += rs[tid + s]; rs2[tid] += rs2[tid + s]; }
        __syncthreads();
    }
    float mean = rs[0] / (float)DM;
    float var  = rs2[0] / (float)DM - mean * mean;
    float inv  = rsqrtf(var + 1e-5f);
    out[row * DM + tid]       = (v0 - mean) * inv * g[tid]       + bt[tid];
    out[row * DM + tid + 256] = (v1 - mean) * inv * g[tid + 256] + bt[tid + 256];
}

// RBF channel: forward/backward fill then 50-center RBF mean -> tok rows.
__launch_bounds__(128)
__global__ void rbf_kernel(const float* __restrict__ xenc, float* __restrict__ tok)
{
    __shared__ float xf[LSEQ], xb[LSEQ];
    const int b = blockIdx.x / 280, c = blockIdx.x % 280;
    const float* s = xenc + (long)b * LSEQ * NCH + (21 + c);
    const int tid = threadIdx.x;
    if (tid == 0) {
        float cur = s[0];
        for (int t = 0; t < LSEQ; ++t) { float v = s[(long)t * NCH]; if (v != 0.f) cur = v; xf[t] = cur; }
    }
    if (tid == 1) {
        float cur = s[(long)(LSEQ - 1) * NCH];
        for (int t = LSEQ - 1; t >= 0; --t) { float v = s[(long)t * NCH]; if (v != 0.f) cur = v; xb[t] = cur; }
    }
    __syncthreads();
    for (int t = tid; t < LSEQ; t += 128) {
        float xi = 0.5f * (xf[t] + xb[t]);
        float acc = 0.f;
        for (int j = 0; j < 50; ++j) {
            float cc = -1.f + 2.f * (float)j / 49.f;
            float d = xi - cc;
            acc += expf(-10.f * d * d);
        }
        tok[((long)b * LTOK + 20 + c) * LSEQ + t] = acc * 0.02f;
    }
}

__global__ void tok_rest_kernel(const float* __restrict__ xenc,
                                const float* __restrict__ xmark, float* __restrict__ tok)
{
    long i = (long)blockIdx.x * blockDim.x + threadIdx.x;
    if (i >= (long)BATCH * 27 * LSEQ) return;
    int t = (int)(i % LSEQ);
    long r = i / LSEQ;
    int rr = (int)(r % 27);
    int b  = (int)(r / 27);
    float v; int row;
    if (rr < 20) { row = rr;              v = xenc[((long)b * LSEQ + t) * NCH + 1 + rr]; }
    else         { row = 300 + (rr - 20); v = xmark[((long)b * LSEQ + t) * 7 + (rr - 20)]; }
    tok[((long)b * LTOK + row) * LSEQ + t] = v;
}

// DFT matrices (len 307, prime) generated directly as f16 GEMM A-operands.
__global__ void dftmat_kernel(_Float16* __restrict__ Cm, _Float16* __restrict__ Sm)
{
    long i = (long)blockIdx.x * blockDim.x + threadIdx.x;
    if (i >= (long)LTOK * LTOK) return;
    int f = (int)(i / LTOK), t = (int)(i % LTOK);
    int r = (int)(((long)f * t) % LTOK); // exact phase via integer mod
    float ang = 6.283185307179586f * (float)r / (float)LTOK;
    Cm[i] = (_Float16)cosf(ang);
    Sm[i] = (_Float16)(-sinf(ang));
}

// Per-channel FFT filter collapses to elementwise scale sigmoid(m)*cos(phi).
__global__ void freq_scale_kernel(float* __restrict__ enc,
                                  const float* __restrict__ fwm, const float* __restrict__ fwp,
                                  const float* __restrict__ fmm, const float* __restrict__ fmp)
{
    long i = (long)blockIdx.x * blockDim.x + threadIdx.x;
    if (i >= (long)BATCH * LTOK * DM) return;
    int ch = (int)(i & (DM - 1));
    float f = 1.f;
    if (ch >= 1 && ch < 13)      { int j = ch - 1;  f = sigm(fwm[j]) * cosf(fwp[j]); }
    else if (ch >= 20)           { int j = ch - 20; f = sigm(fmm[j]) * cosf(fmp[j]); }
    enc[i] *= f;
}

// ilt basis + gate, writes X rows 0..306.
__launch_bounds__(256)
__global__ void ilt_kernel(const float* __restrict__ mapped,
                           const float* __restrict__ pA, const float* __restrict__ pS,
                           const float* __restrict__ pW, const float* __restrict__ pPhi,
                           float* __restrict__ X)
{
    __shared__ float rs[256];
    const int row = blockIdx.x;              // 0..B*307-1
    const int b = row / LTOK, t = row % LTOK;
    const float* mp = mapped + (long)row * DM;
    const int tid = threadIdx.x;
    float v0 = mp[tid], v1 = mp[tid + 256];
    rs[tid] = v0 + v1;
    __syncthreads();
    for (int s = 128; s > 0; s >>= 1) { if (tid < s) rs[tid] += rs[tid + s]; __syncthreads(); }
    const float gate = sigm(rs[0] / (float)DM);
    const float tf = (float)t;
    float* xp = X + ((long)b * LENC + t) * DM;
    for (int d = tid; d < DM; d += 256) {
        float sa = log1pf(expf(pA[d]));
        float sg = log1pf(expf(pS[d]));
        float sw = log1pf(expf(pW[d]));
        float ilt = sa * expf(-sg * tf) * cosf(sw * tf + pPhi[d]);
        xp[d] = gate * ilt + (1.f - gate) * mp[d];
    }
}

__global__ void extract_endo(const float* __restrict__ xenc, float* __restrict__ out)
{
    int i = blockIdx.x * blockDim.x + threadIdx.x;
    if (i >= BATCH * LSEQ) return;
    int b = i / LSEQ, t = i % LSEQ;
    out[i] = xenc[((long)b * LSEQ + t) * NCH];
}

// positional embedding for n=1: even dims sin(0)=0, odd dims cos(0)=1.
__global__ void pe_kernel(float* __restrict__ pe)
{
    int i = blockIdx.x * blockDim.x + threadIdx.x;
    if (i < DM) pe[i] = (i & 1) ? 1.f : 0.f;
}

__global__ void fill_glb(const float* __restrict__ glb, float* __restrict__ E)
{
    int i = blockIdx.x * blockDim.x + threadIdx.x;
    if (i >= BATCH * DM) return;
    int b = i / DM, d = i % DM;
    E[(long)b * (2 * DM) + DM + d] = glb[d];
}

__global__ void copy_endo(const float* __restrict__ E, float* __restrict__ X)
{
    int i = blockIdx.x * blockDim.x + threadIdx.x;
    if (i >= BATCH * DM) return;
    int b = i / DM, d = i % DM;
    X[((long)b * LENC + 307) * DM + d] = E[(long)b * (2 * DM) + d];
}

// final projection + transpose: out (B, 7, 308)
__global__ void proj_kernel(const float* __restrict__ X, const float* __restrict__ Wp,
                            const float* __restrict__ bp, float* __restrict__ out)
{
    long i = (long)blockIdx.x * blockDim.x + threadIdx.x;
    if (i >= (long)BATCH * LENC * PLEN) return;
    int p = (int)(i % PLEN);
    long r = i / PLEN;
    int l = (int)(r % LENC);
    int b = (int)(r / LENC);
    const float* xp = X + ((long)b * LENC + l) * DM;
    float a = bp[p];
    for (int d = 0; d < DM; ++d) a += xp[d] * Wp[d * PLEN + p];
    out[((long)b * PLEN + p) * LENC + l] = a;
}

// ---------------------------------------------------------------------------
// Host side
// ---------------------------------------------------------------------------
struct LayerP {
    const float *Wq, *bq, *Wk, *bk, *Wv, *bv, *Wo, *bo;
    const float *W1, *bf1, *W2, *bf2, *g1, *be1, *g2, *be2;
};

static LayerP layer_from(void* const* d, int b)
{
    LayerP L;
    L.Wq  = (const float*)d[b + 0];  L.bq  = (const float*)d[b + 1];
    L.Wk  = (const float*)d[b + 2];  L.bk  = (const float*)d[b + 3];
    L.Wv  = (const float*)d[b + 4];  L.bv  = (const float*)d[b + 5];
    L.Wo  = (const float*)d[b + 6];  L.bo  = (const float*)d[b + 7];
    L.W1  = (const float*)d[b + 8];  L.bf1 = (const float*)d[b + 9];
    L.W2  = (const float*)d[b + 10]; L.bf2 = (const float*)d[b + 11];
    L.g1  = (const float*)d[b + 12]; L.be1 = (const float*)d[b + 13];
    L.g2  = (const float*)d[b + 14]; L.be2 = (const float*)d[b + 15];
    return L;
}

// A: f32 [M,K] (packed to f16) or pre-packed f16 (A16pre). W: f32 [K,N] per
// batch, packed TRANSPOSED to f16 [N,K]. batch via blockIdx.z.
static void gemm(hipStream_t st,
                 const float* A, const _Float16* A16pre, long sA,
                 const float* W, long sWin,
                 const float* bias, float* C, long sC, int ldC,
                 int M, int N, int K, int act, int batch,
                 _Float16* PA, _Float16* PW)
{
    const _Float16* Ause = A16pre;
    if (!A16pre) {
        long totalA = (long)M * K;
        pack_rm<<<(int)((totalA + 255) / 256), 256, 0, st>>>(A, PA, totalA);
        Ause = PA;
    }
    long sInW = sWin ? sWin : (long)K * N;
    long totalW = (long)K * N * batch;
    pack_tr<<<(int)((totalW + 255) / 256), 256, 0, st>>>(W, PW, K, N, sInW, (long)K * N, totalW);
    dim3 g((unsigned)((M + 127) / 128), (unsigned)(N / 64), (unsigned)batch);
    gemm_wmma<<<g, 256, 0, st>>>(Ause, sA, PW, (long)K * N, bias, C, sC, ldC, M, N, K, act);
}

static void enc_layer_run(hipStream_t st, float* Xb, int rows, int L, int Bc, const LayerP& p,
                          float* Qb, float* Kb, float* Vb, float* Ab, float* Yb, float* H1,
                          _Float16* PA, _Float16* PW)
{
    gemm(st, Xb, nullptr, 0, p.Wq, 0, p.bq, Qb, 0, DM, rows, DM, DM, 0, 1, PA, PW);
    gemm(st, Xb, nullptr, 0, p.Wk, 0, p.bk, Kb, 0, DM, rows, DM, DM, 0, 1, PA, PW);
    gemm(st, Xb, nullptr, 0, p.Wv, 0, p.bv, Vb, 0, DM, rows, DM, DM, 0, 1, PA, PW);
    size_t shm = (size_t)(3 * L * 64 + 64 * 64 + 4 * L + 4 * 64 + 64) * sizeof(float);
    flow_attn<<<dim3(Bc * NHEAD), 256, shm, st>>>(Qb, Kb, Vb, Ab, L);
    gemm(st, Ab, nullptr, 0, p.Wo, 0, p.bo, Yb, 0, DM, rows, DM, DM, 0, 1, PA, PW);
    ln_kernel<<<rows, 256, 0, st>>>(Xb, Yb, p.g1, p.be1, Xb);
    gemm(st, Xb, nullptr, 0, p.W1, 0, p.bf1, H1, 0, DFFN, rows, DFFN, DM, 1, 1, PA, PW); // GELU
    gemm(st, H1, nullptr, 0, p.W2, 0, p.bf2, Yb, 0, DM, rows, DM, DFFN, 0, 1, PA, PW);
    ln_kernel<<<rows, 256, 0, st>>>(Xb, Yb, p.g2, p.be2, Xb);
}

// Input leaf order assumption: setup_inputs() dict insertion order, params
// flattened depth-first in insertion order (73 leaves).
enum {
    I_XENC = 0, I_XMARK = 1, I_XDEC = 2, I_XMARKDEC = 3,
    I_EN_W = 4, I_GLB = 5, I_ENL0 = 6, /* 16 leaves */
    I_EN_NG = 22, I_EN_NB = 23,
    I_FWM = 24, I_FWP = 25, I_FMM = 26, I_FMP = 27,
    I_FFTW = 28, I_ILTW = 29, I_ILTB = 30, I_ILTA = 31, I_ILTS = 32,
    I_ILTOM = 33, I_ILTPHI = 34,
    I_EMBW = 35, I_EMBB = 36,
    I_L0 = 37, I_L1 = 53,
    I_NG = 69, I_NB = 70, I_PROJW = 71, I_PROJB = 72
};

extern "C" void kernel_launch(void* const* d_in, const int* in_sizes, int n_in,
                              void* d_out, int out_size, void* d_ws, size_t ws_size,
                              hipStream_t stream)
{
    (void)in_sizes; (void)n_in; (void)out_size; (void)ws_size;
    const float* x_enc  = (const float*)d_in[I_XENC];
    const float* x_mark = (const float*)d_in[I_XMARK];
    const float* en_W   = (const float*)d_in[I_EN_W];
    const float* glb    = (const float*)d_in[I_GLB];
    const float* en_ng  = (const float*)d_in[I_EN_NG];
    const float* en_nb  = (const float*)d_in[I_EN_NB];
    const float* fwm    = (const float*)d_in[I_FWM];
    const float* fwp    = (const float*)d_in[I_FWP];
    const float* fmm    = (const float*)d_in[I_FMM];
    const float* fmp    = (const float*)d_in[I_FMP];
    const float* fftW   = (const float*)d_in[I_FFTW];
    const float* iltW   = (const float*)d_in[I_ILTW];
    const float* iltB   = (const float*)d_in[I_ILTB];
    const float* iltA   = (const float*)d_in[I_ILTA];
    const float* iltS   = (const float*)d_in[I_ILTS];
    const float* iltOm  = (const float*)d_in[I_ILTOM];
    const float* iltPhi = (const float*)d_in[I_ILTPHI];
    const float* embW   = (const float*)d_in[I_EMBW];
    const float* embB   = (const float*)d_in[I_EMBB];
    const float* ng     = (const float*)d_in[I_NG];
    const float* nb     = (const float*)d_in[I_NB];
    const float* projW  = (const float*)d_in[I_PROJW];
    const float* projB  = (const float*)d_in[I_PROJB];
    LayerP enl0 = layer_from(d_in, I_ENL0);
    LayerP l0   = layer_from(d_in, I_L0);
    LayerP l1   = layer_from(d_in, I_L1);
    float* outp = (float*)d_out;

    // --- workspace layout (float units); ~185 MB total ---
    float* wsf = (float*)d_ws;
    size_t off = 0;
    auto alloc = [&](size_t nfloats) {
        float* p = wsf + off; off += (nfloats + 255) & ~(size_t)255; return p;
    };
    auto allocH = [&](size_t nhalfs) { return (_Float16*)alloc((nhalfs + 1) / 2); };

    float* endoA  = alloc((size_t)BATCH * LSEQ);
    float* tok    = alloc((size_t)BATCH * LTOK * LSEQ);
    float* enc    = alloc((size_t)BATCH * LTOK * DM);
    float* fr     = alloc((size_t)BATCH * LTOK * 2 * DM);
    float* filt   = alloc((size_t)BATCH * LTOK * DM);
    float* mapped = alloc((size_t)BATCH * LTOK * DM);
    float* X      = alloc((size_t)BATCH * LENC * DM);
    float* Qb     = alloc((size_t)BATCH * LENC * DM);
    float* Kb     = alloc((size_t)BATCH * LENC * DM);
    float* Vb     = alloc((size_t)BATCH * LENC * DM);
    float* Ab     = alloc((size_t)BATCH * LENC * DM);
    float* Yb     = alloc((size_t)BATCH * LENC * DM);
    float* H1     = alloc((size_t)BATCH * LENC * DFFN);
    float* pe     = alloc(DM);
    float* E      = alloc((size_t)BATCH * 2 * DM);
    _Float16* Cm16 = allocH((size_t)LTOK * LTOK);
    _Float16* Sm16 = allocH((size_t)LTOK * LTOK);
    _Float16* PA   = allocH((size_t)BATCH * LENC * DFFN);     // max A f16
    _Float16* PW   = allocH((size_t)BATCH * LTOK * DM);       // max W f16 (DFT batch)

    // --- constants ---
    dftmat_kernel<<<(LTOK * LTOK + 255) / 256, 256, 0, stream>>>(Cm16, Sm16);
    pe_kernel<<<2, 256, 0, stream>>>(pe);

    // --- endogenous path ---
    extract_endo<<<(BATCH * LSEQ + 255) / 256, 256, 0, stream>>>(x_enc, endoA);
    gemm(stream, endoA, nullptr, 0, en_W, 0, pe, E, 0, 2 * DM, BATCH, DM, LSEQ, 0, 1, PA, PW);
    fill_glb<<<(BATCH * DM + 255) / 256, 256, 0, stream>>>(glb, E);
    enc_layer_run(stream, E, BATCH * 2, 2, BATCH, enl0, Qb, Kb, Vb, Ab, Yb, H1, PA, PW);
    ln_kernel<<<BATCH * 2, 256, 0, stream>>>(E, nullptr, en_ng, en_nb, E);
    copy_endo<<<(BATCH * DM + 255) / 256, 256, 0, stream>>>(E, X); // X row 307

    // --- exogenous path: RBF + tok build + embedding GEMM ---
    rbf_kernel<<<BATCH * 280, 128, 0, stream>>>(x_enc, tok);
    tok_rest_kernel<<<(int)(((long)BATCH * 27 * LSEQ + 255) / 256), 256, 0, stream>>>(x_enc, x_mark, tok);
    gemm(stream, tok, nullptr, 0, embW, 0, embB, enc, 0, DM, BATCH * LTOK, DM, LSEQ, 0, 1, PA, PW);

    // per-channel FFT filter -> elementwise scale
    freq_scale_kernel<<<(int)(((long)BATCH * LTOK * DM + 255) / 256), 256, 0, stream>>>(
        enc, fwm, fwp, fmm, fmp);

    // fft_ilt: DFT(len 307) as 2 batched GEMMs -> fr = [real | imag]
    gemm(stream, nullptr, Cm16, 0, enc, (long)LTOK * DM, nullptr,
         fr,      (long)LTOK * 2 * DM, 2 * DM, LTOK, DM, LTOK, 0, BATCH, PA, PW);
    gemm(stream, nullptr, Sm16, 0, enc, (long)LTOK * DM, nullptr,
         fr + DM, (long)LTOK * 2 * DM, 2 * DM, LTOK, DM, LTOK, 0, BATCH, PA, PW);
    gemm(stream, fr,   nullptr, 0, fftW, 0, nullptr, filt,   0, DM, BATCH * LTOK, DM, 2 * DM, 0, 1, PA, PW);
    gemm(stream, filt, nullptr, 0, iltW, 0, iltB,    mapped, 0, DM, BATCH * LTOK, DM, DM,     0, 1, PA, PW);
    ilt_kernel<<<BATCH * LTOK, 256, 0, stream>>>(mapped, iltA, iltS, iltOm, iltPhi, X); // rows 0..306

    // --- main encoder (2 layers) + final LN + projection ---
    enc_layer_run(stream, X, BATCH * LENC, LENC, BATCH, l0, Qb, Kb, Vb, Ab, Yb, H1, PA, PW);
    enc_layer_run(stream, X, BATCH * LENC, LENC, BATCH, l1, Qb, Kb, Vb, Ab, Yb, H1, PA, PW);
    ln_kernel<<<BATCH * LENC, 256, 0, stream>>>(X, nullptr, ng, nb, X);
    proj_kernel<<<(int)(((long)BATCH * LENC * PLEN + 255) / 256), 256, 0, stream>>>(
        X, projW, projB, outp);
}
